// GMLQuantConvBase_34291018891375
// MI455X (gfx1250) — matrix operand order repaired and checked by reading, not compiled, not run on hardware
//
#include <hip/hip_runtime.h>

typedef __attribute__((ext_vector_type(16))) _Float16 v16h;
typedef __attribute__((ext_vector_type(8)))  float    v8f;

#define BATCH 16
#define CIN   32
#define COUT  32
#define HH    256
#define WW    256
#define HTILE 4
#define WTILE 32
#define XROWS (HTILE + 2)   // 6
#define XCOLS (WTILE + 2)   // 34
#define CPAD  40            // 80 B row: 16B-aligned runs + conflict-free (20 DW stride)

union HU { v16h h; uint4 q[2]; };

__global__ __launch_bounds__(256)
void qconv_wmma_f16(const int* __restrict__ x,
                    const float* __restrict__ wgt,
                    const float* __restrict__ bias,
                    float* __restrict__ out) {
    // [kh*3+kw][cout][cin(pad 40)], weights pre-scaled by IN_SCALE, as f16
    __shared__ __align__(16) _Float16 wlds[9 * COUT * CPAD];        // 23040 B
    // [row][col][cin(pad 40)], integer-valued f16 input tile with halo
    __shared__ __align__(16) _Float16 xlds[XROWS * XCOLS * CPAD];   // 16320 B

    const int tid  = threadIdx.x;
    const int bid  = blockIdx.x;
    const int wblk = bid & 7;          // WW/WTILE = 8
    const int hblk = (bid >> 3) & 63;  // HH/HTILE = 64
    const int b    = bid >> 9;         // BATCH

    // ---- stage weights: iterate in GLOBAL order (coalesced reads),
    //      scatter into [khw][cout][cin] LDS, * IN_SCALE ----
    for (int j = tid; j < COUT * CIN * 9; j += 256) {
        int khw  = j % 9;              // fastest in global OIHW layout
        int t    = j / 9;
        int cin  = t & 31;
        int cout = t >> 5;
        float v  = wgt[j] * 0.05f;
        wlds[(khw * COUT + cout) * CPAD + cin] = (_Float16)v;
    }

    // ---- stage input tile: iterate with col (ww) fastest -> coalesced
    //      global reads; scatter into [row][col][cin] LDS ----
    const int h0 = hblk * HTILE - 1;
    const int w0 = wblk * WTILE - 1;
    for (int i = tid; i < XROWS * CIN * XCOLS; i += 256) {
        int col = i % XCOLS;           // fastest: matches global ww stride 1
        int t   = i / XCOLS;
        int cin = t & 31;
        int row = t >> 5;
        int hh  = h0 + row;
        int ww  = w0 + col;
        _Float16 v = (_Float16)0.f;
        if (hh >= 0 && hh < HH && ww >= 0 && ww < WW)
            v = (_Float16)(float)x[((b * CIN + cin) * HH + hh) * WW + ww];
        xlds[(row * XCOLS + col) * CPAD + cin] = v;
    }
    __syncthreads();

    // ---- per-wave 16x16x32 WMMA implicit GEMM ----
    const int lane = tid & 31;
    const int wv   = tid >> 5;
    const int r    = wv >> 1;      // output row within tile: 0..3
    const int s    = wv & 1;       // 16-wide span: 0..1
    const int hi   = lane >> 4;    // lane half
    const int lo   = lane & 15;

    v8f c0 = {}; // cout 0..15
    v8f c1 = {}; // cout 16..31

#pragma unroll
    for (int kh = 0; kh < 3; ++kh) {
#pragma unroll
        for (int kw = 0; kw < 3; ++kw) {
            const _Float16* wp = &wlds[(kh * 3 + kw) * COUT * CPAD];
            const _Float16* xp = &xlds[((r + kh) * XCOLS + (s * 16 + lo + kw)) * CPAD];

            HU a0, a1, bm;
            // A layout (16x32 f16): lane half hi holds K runs [hi*8, hi*8+7]
            // and [16+hi*8, 16+hi*8+7]; each run is one aligned 16B LDS load.
            a0.q[0] = *(const uint4*)&wp[lo * CPAD + hi * 8];
            a0.q[1] = *(const uint4*)&wp[lo * CPAD + hi * 8 + 16];
            a1.q[0] = *(const uint4*)&wp[(lo + 16) * CPAD + hi * 8];
            a1.q[1] = *(const uint4*)&wp[(lo + 16) * CPAD + hi * 8 + 16];
            // B layout (32x16 f16): lane half hi holds K runs [hi*8, hi*8+7]
            // and [16+hi*8, 16+hi*8+7]
            bm.q[0] = *(const uint4*)&xp[hi * 8];
            bm.q[1] = *(const uint4*)&xp[hi * 8 + 16];

            c0 = __builtin_amdgcn_wmma_f32_16x16x32_f16(
                     false, a0.h, false, bm.h, (short)0, c0, false, false);
            c1 = __builtin_amdgcn_wmma_f32_16x16x32_f16(
                     false, a1.h, false, bm.h, (short)0, c1, false, false);
        }
    }

    // ---- epilogue: bias + ReLU + requantize, write float-encoded int8 ----
    const int hout = hblk * HTILE + r;
    const int wout = wblk * WTILE + s * 16 + lo;
#pragma unroll
    for (int i = 0; i < 8; ++i) {
        int m = i + hi * 8;                 // row of the 16x16 C tile
        float y0 = c0[i] + bias[m];
        float y1 = c1[i] + bias[m + 16];
        y0 = fmaxf(y0, 0.f);
        y1 = fmaxf(y1, 0.f);
        float q0 = fminf(fmaxf(rintf(y0 * 10.0f), -128.f), 127.f);
        float q1 = fminf(fmaxf(rintf(y1 * 10.0f), -128.f), 127.f);
        out[((b * COUT + m)      * HH + hout) * WW + wout] = q0;
        out[((b * COUT + m + 16) * HH + hout) * WW + wout] = q1;
    }
}

extern "C" void kernel_launch(void* const* d_in, const int* in_sizes, int n_in,
                              void* d_out, int out_size, void* d_ws, size_t ws_size,
                              hipStream_t stream) {
    const int*   x    = (const int*)d_in[0];
    const float* wgt  = (const float*)d_in[1];
    const float* bias = (const float*)d_in[2];
    float*       out  = (float*)d_out;

    // grid = BATCH * (HH/HTILE) * (WW/WTILE) = 16 * 64 * 8 = 8192
    dim3 grid(8192), block(256);
    hipLaunchKernelGGL(qconv_wmma_f16, grid, block, 0, stream, x, wgt, bias, out);
}
